// VectorQuantizer_635655160473
// MI455X (gfx1250) — matrix operand order repaired and checked by reading, not compiled, not run on hardware
//
#include <hip/hip_runtime.h>
#include <hip/hip_bf16.h>

typedef __attribute__((ext_vector_type(16))) _Float16 v16h;
typedef __attribute__((ext_vector_type(8)))  _Float16 v8h;
typedef __attribute__((ext_vector_type(8)))  float    v8f;

#define NVEC   65536      // 16*64*64 flattened pixels
#define DIM    64
#define KCODES 512
#define TILE_M 128        // rows per workgroup (4 waves x 32 rows)
#define THREADS 128
#define XH_S   72         // Xh row stride (halves): 144B, 16B aligned, bank-spread
#define XF_S   65         // Xf row stride (floats): conflict-free column access

// d_ws layout (bytes):
//   [0, 65536)        : codebook as f16  (512*64*2)
//   [65536, 67584)    : ||c||^2 fp32     (512*4)
//   [67584, 69632)    : histogram u32    (512*4)
//   [69632, 69636)    : SSE accumulator  (1*4)
#define WS_CBH  0
#define WS_C2   65536
#define WS_HIST 67584
#define WS_SSE  69632

static __device__ __forceinline__ v16h cat16(v8h a, v8h b) {
    return __builtin_shufflevector(a, b, 0, 1, 2, 3, 4, 5, 6, 7,
                                         8, 9, 10, 11, 12, 13, 14, 15);
}

// ---------------------------------------------------------------------------
// Prep: f16 codebook copy, ||c||^2, zero hist + sse.  <<<512, 64>>>
// ---------------------------------------------------------------------------
__global__ void vq_prep(const float* __restrict__ cb,
                        _Float16* __restrict__ cbh,
                        float* __restrict__ c2,
                        unsigned* __restrict__ hist,
                        float* __restrict__ sse) {
    const int c = blockIdx.x;
    const int d = threadIdx.x;
    __shared__ float s[DIM];
    float v = cb[c * DIM + d];
    cbh[c * DIM + d] = (_Float16)v;
    s[d] = v * v;
    __syncthreads();
    for (int o = 32; o > 0; o >>= 1) {
        if (d < o) s[d] += s[d + o];
        __syncthreads();
    }
    if (d == 0) { c2[c] = s[0]; hist[c] = 0u; }
    if (c == 0 && d == 1) *sse = 0.0f;
}

// ---------------------------------------------------------------------------
// Main: WMMA distance argmin + gather + loss partials.  <<<512, 128>>>
// ---------------------------------------------------------------------------
__global__ void __launch_bounds__(THREADS)
vq_main(const float* __restrict__ x,        // [16, 64, 64, 64] (B,C,H,W)
        const float* __restrict__ cb,       // [512, 64] fp32
        const _Float16* __restrict__ cbh,   // [512, 64] f16
        const float* __restrict__ c2,       // [512]
        float* __restrict__ out,            // [16, 64, 64, 64]
        unsigned* __restrict__ hist,
        float* __restrict__ sse) {
    __shared__ __align__(16) _Float16 Xh[TILE_M * XH_S];
    __shared__ float Xf[TILE_M * XF_S];
    __shared__ int   idxLds[TILE_M];
    __shared__ float red[THREADS];

    const int t   = threadIdx.x;
    const int n0  = blockIdx.x * TILE_M;      // first flattened pixel
    const int b   = n0 >> 12;                 // n = b*4096 + h*64 + w
    const int hw0 = n0 & 4095;                // multiple of 128, within one batch
    const float* xb = x + ((size_t)b << 18);  // b * 64 * 4096

    // ---- stage X tile: thread t owns pixel row j = t, loops channels ----
    {
        const int j = t;
#pragma unroll 8
        for (int d = 0; d < DIM; ++d) {
            const float v = xb[(size_t)d * 4096 + hw0 + j];
            Xf[j * XF_S + d] = v;
            Xh[j * XH_S + d] = (_Float16)v;
        }
    }
    __syncthreads();

    // ---- per-wave 32-row strip pair (2 x 16-row WMMA strips) ----
    const int wave = t >> 5, lane = t & 31;
    const int r0   = wave * 32;
    const int half = lane >> 4;   // 0: lanes 0-15, 1: lanes 16-31
    const int m    = lane & 15;

    // A fragments (ISA 16-bit A 16x32 layout), reused across all code tiles.
    v16h A[2][2];
#pragma unroll
    for (int s = 0; s < 2; ++s) {
        const _Float16* row = &Xh[(r0 + s * 16 + m) * XH_S];
        const int ko = half * 8;
        A[s][0] = cat16(*(const v8h*)(row + ko),      *(const v8h*)(row + 16 + ko));
        A[s][1] = cat16(*(const v8h*)(row + 32 + ko), *(const v8h*)(row + 48 + ko));
    }

    float bestV[2][8];
    int   bestI[2][8];
#pragma unroll
    for (int s = 0; s < 2; ++s)
#pragma unroll
        for (int r = 0; r < 8; ++r) { bestV[s][r] = 3.4e38f; bestI[s][r] = 0; }

    for (int ct = 0; ct < 32; ++ct) {
        const int code = ct * 16 + m;         // this lane's column (code id)
        const _Float16* crow = cbh + code * DIM + half * 8;
        // Issue all four 16B B-fragment loads before any use (one clause).
        const v8h b00 = *(const v8h*)(crow);
        const v8h b01 = *(const v8h*)(crow + 16);
        const v8h b10 = *(const v8h*)(crow + 32);
        const v8h b11 = *(const v8h*)(crow + 48);
        const float cc = c2[code];
        if (ct < 31)  // gfx1250 global_prefetch_b8: pull next tile's rows
            __builtin_prefetch((const void*)(crow + 16 * DIM), 0, 1);
        const v16h B0 = cat16(b00, b01);
        const v16h B1 = cat16(b10, b11);
#pragma unroll
        for (int s = 0; s < 2; ++s) {
            v8f acc = {0.f, 0.f, 0.f, 0.f, 0.f, 0.f, 0.f, 0.f};
            acc = __builtin_amdgcn_wmma_f32_16x16x32_f16(false, A[s][0], false, B0,
                                                         (short)0, acc, false, false);
            acc = __builtin_amdgcn_wmma_f32_16x16x32_f16(false, A[s][1], false, B1,
                                                         (short)0, acc, false, false);
#pragma unroll
            for (int r = 0; r < 8; ++r) {
                const float sc = cc - 2.0f * acc[r];   // ||c||^2 - 2 x.c
                if (sc < bestV[s][r]) { bestV[s][r] = sc; bestI[s][r] = code; }
            }
        }
    }

    // min-reduce (val,idx) across the 16 lanes of each half-wave.
#pragma unroll
    for (int off = 1; off <= 8; off <<= 1) {
#pragma unroll
        for (int s = 0; s < 2; ++s)
#pragma unroll
            for (int r = 0; r < 8; ++r) {
                const float v2 = __shfl_xor(bestV[s][r], off, 32);
                const int   i2 = __shfl_xor(bestI[s][r], off, 32);
                if (v2 < bestV[s][r] || (v2 == bestV[s][r] && i2 < bestI[s][r])) {
                    bestV[s][r] = v2; bestI[s][r] = i2;
                }
            }
    }
    if (m == 0) {
#pragma unroll
        for (int s = 0; s < 2; ++s) {
            const int rbase = r0 + s * 16 + half * 8;  // lanes<16 -> M 0-7, else 8-15
#pragma unroll
            for (int r = 0; r < 8; ++r) idxLds[rbase + r] = bestI[s][r];
        }
    }
    __syncthreads();

    // ---- gather fp32 codebook rows, write output, accumulate SSE ----
    float local = 0.0f;
    {
        const int j = t;
        const int ci = idxLds[j];
        const float* qrow = cb + ci * DIM;
        const size_t obase = ((size_t)(b * 64) << 12) + hw0 + j;
#pragma unroll 4
        for (int d = 0; d < DIM; d += 4) {
            const float4 q = *(const float4*)(qrow + d);
            out[obase + ((size_t)(d + 0) << 12)] = q.x;
            out[obase + ((size_t)(d + 1) << 12)] = q.y;
            out[obase + ((size_t)(d + 2) << 12)] = q.z;
            out[obase + ((size_t)(d + 3) << 12)] = q.w;
            float df;
            df = q.x - Xf[j * XF_S + d + 0]; local += df * df;
            df = q.y - Xf[j * XF_S + d + 1]; local += df * df;
            df = q.z - Xf[j * XF_S + d + 2]; local += df * df;
            df = q.w - Xf[j * XF_S + d + 3]; local += df * df;
        }
    }
    red[t] = local;
    __syncthreads();
    for (int s = THREADS / 2; s > 0; s >>= 1) {
        if (t < s) red[t] += red[t + s];
        __syncthreads();
    }
    if (t == 0) atomicAdd(sse, red[0]);
    if (t < TILE_M) atomicAdd(&hist[idxLds[t]], 1u);
}

// ---------------------------------------------------------------------------
// Finalize: loss + perplexity.  <<<1, 512>>>
// ---------------------------------------------------------------------------
__global__ void vq_finalize(const unsigned* __restrict__ hist,
                            const float* __restrict__ sse,
                            float* __restrict__ tail) {
    __shared__ float red[KCODES];
    const int t = threadIdx.x;
    const float p = (float)hist[t] * (1.0f / (float)NVEC);
    red[t] = p * logf(p + 1e-10f);
    __syncthreads();
    for (int s = KCODES / 2; s > 0; s >>= 1) {
        if (t < s) red[t] += red[t + s];
        __syncthreads();
    }
    if (t == 0) {
        tail[0] = 1.25f * (*sse) / (float)((size_t)NVEC * DIM);  // q + 0.25*e
        tail[1] = expf(-red[0]);
    }
}

extern "C" void kernel_launch(void* const* d_in, const int* in_sizes, int n_in,
                              void* d_out, int out_size, void* d_ws, size_t ws_size,
                              hipStream_t stream) {
    (void)in_sizes; (void)n_in; (void)out_size; (void)ws_size;
    const float* x  = (const float*)d_in[0];
    const float* cb = (const float*)d_in[1];
    float* out = (float*)d_out;

    char* ws = (char*)d_ws;
    _Float16* cbh  = (_Float16*)(ws + WS_CBH);
    float*    c2   = (float*)(ws + WS_C2);
    unsigned* hist = (unsigned*)(ws + WS_HIST);
    float*    sse  = (float*)(ws + WS_SSE);

    vq_prep<<<KCODES, DIM, 0, stream>>>(cb, cbh, c2, hist, sse);
    vq_main<<<NVEC / TILE_M, THREADS, 0, stream>>>(x, cb, cbh, c2, out, hist, sse);
    vq_finalize<<<1, KCODES, 0, stream>>>(hist, sse, out + (size_t)NVEC * DIM);
}
